// bidirLSTM_Layer_52939766891148
// MI455X (gfx1250) — compile-verified, matrix operand order checked by reading, not run on hardware
//
#include <hip/hip_runtime.h>

// ---------------------------------------------------------------------------
// Types for CDNA5 WMMA (gfx1250, wave32)
// ---------------------------------------------------------------------------
typedef __attribute__((ext_vector_type(16))) __bf16 bf16x16;
typedef __attribute__((ext_vector_type(8)))  float  floatx8;

union Frag {
    bf16x16 v;
    __bf16  e[16];
    uint4   q[2];
};

__device__ __forceinline__ float sigmoidf_(float x) {
    return 1.0f / (1.0f + __expf(-x));
}
__device__ __forceinline__ float tanhf_(float x) {
    float xc = fminf(fmaxf(x, -15.0f), 15.0f);
    float e  = __expf(2.0f * xc);
    return (e - 1.0f) / (e + 1.0f);
}

#define WMMA_BF16(A, B, C) \
    __builtin_amdgcn_wmma_f32_16x16x32_bf16(false, (A), false, (B), (short)0, (C), false, false)

// ---------------------------------------------------------------------------
// Sizes (fixed by the reference): B=16, S=4096, H=F=256, 4F=1024
// ---------------------------------------------------------------------------
#define SEQ   4096
#define BATCH 16
#define HID   256
#define GATE  1024   // 4*F

// ---------------------------------------------------------------------------
// Kernel 0: stage x [16][4096][256] f32 -> bf16 (written once, read as b128
// fragments by the projection GEMM; halves A-side bytes, removes converts
// from the GEMM inner loop)
// ---------------------------------------------------------------------------
__global__ void pack_x_kernel(const float* __restrict__ x,
                              __bf16* __restrict__ xbf) {
    const size_t i4 = (size_t)blockIdx.x * blockDim.x + threadIdx.x;  // 4 elems/thread
    const float4 f = ((const float4*)x)[i4];
    union { __bf16 b[4]; uint2 u; } p;
    p.b[0] = (__bf16)f.x; p.b[1] = (__bf16)f.y;
    p.b[2] = (__bf16)f.z; p.b[3] = (__bf16)f.w;
    ((uint2*)xbf)[i4] = p.u;
}

// ---------------------------------------------------------------------------
// Kernel 1: pack the four [256][1024] f32 weight matrices into bf16 [1024][256]
// (transposed, so B-fragments become contiguous 16-byte chunks)
// ---------------------------------------------------------------------------
__global__ void pack_weights_kernel(const float* __restrict__ Wi_fw,
                                    const float* __restrict__ Wh_fw,
                                    const float* __restrict__ Wi_rv,
                                    const float* __restrict__ Wh_rv,
                                    __bf16* __restrict__ WiT_fw,
                                    __bf16* __restrict__ WhT_fw,
                                    __bf16* __restrict__ WiT_rv,
                                    __bf16* __restrict__ WhT_rv) {
    int idx = blockIdx.x * blockDim.x + threadIdx.x;     // 0 .. 4*1024*256-1
    int w   = idx >> 18;
    int rem = idx & 262143;
    int n   = rem >> 8;       // 0..1023
    int k   = rem & 255;      // 0..255
    const float* s = (w == 0) ? Wi_fw : (w == 1) ? Wh_fw : (w == 2) ? Wi_rv : Wh_rv;
    __bf16* d = (w == 0) ? WiT_fw : (w == 1) ? WhT_fw : (w == 2) ? WiT_rv : WhT_rv;
    d[(size_t)n * 256 + k] = (__bf16)s[(size_t)k * 1024 + n];
}

// ---------------------------------------------------------------------------
// Kernel 2: pre-projection GEMM  xp[s,b,:] = xbf[b, s_eff, :] @ Wi  (bf16 WMMA)
// grid = (8 col-blocks, 4096 time steps, 2 directions), block = 256 (8 waves)
// each wave produces one 16x16 tile (M=batch, N=gate cols), K=256 -> 8 WMMAs.
// Inner loop is pure global_load_b128 + v_wmma.
// ---------------------------------------------------------------------------
__global__ void __launch_bounds__(256)
xproj_gemm_kernel(const __bf16* __restrict__ xbf,
                  const __bf16* __restrict__ WiT_fw,
                  const __bf16* __restrict__ WiT_rv,
                  __bf16* __restrict__ xp_fw,
                  __bf16* __restrict__ xp_rv) {
    const int dir = blockIdx.z;
    const __bf16* WiT = dir ? WiT_rv : WiT_fw;
    __bf16*       xp  = dir ? xp_rv  : xp_fw;

    const int s     = blockIdx.y;
    const int s_eff = dir ? (SEQ - 1 - s) : s;

    const int wave = threadIdx.x >> 5;
    const int lane = threadIdx.x & 31;
    const int hi   = (lane >> 4) & 1;      // lane group
    const int lm   = lane & 15;

    const int n = blockIdx.x * 128 + wave * 16 + lm;     // gate column for B/C frag

    const uint4* xrow = (const uint4*)(xbf + ((size_t)lm * SEQ + s_eff) * HID); // 32 uint4
    const uint4* wrow = (const uint4*)(WiT + (size_t)n * HID);                  // 32 uint4

    floatx8 acc = {0.f, 0.f, 0.f, 0.f, 0.f, 0.f, 0.f, 0.f};

#pragma unroll
    for (int kt = 0; kt < 8; ++kt) {
        Frag a;
        a.q[0] = xrow[kt * 4 + hi];
        a.q[1] = xrow[kt * 4 + hi + 2];

        Frag b;
        b.q[0] = wrow[kt * 4 + hi];
        b.q[1] = wrow[kt * 4 + hi + 2];

        acc = WMMA_BF16(a.v, b.v, acc);
    }

#pragma unroll
    for (int r = 0; r < 8; ++r) {
        const int m = r + hi * 8;            // batch row
        xp[((size_t)s * BATCH + m) * GATE + n] = (__bf16)acc[r];
    }
}

// ---------------------------------------------------------------------------
// Kernel 3: persistent LSTM scan. grid = 2 (directions), block = 1024 (32 waves)
// Each step: z = h @ Wh (WMMA) + xp[t] + b  ->  gates  ->  c,h update in LDS.
// Wave w owns gate-column tiles 2w, 2w+1. Tile0's B-frags stay in VGPRs,
// tile1's B-frags stream from global (L2-resident; same address every step).
// ---------------------------------------------------------------------------
__global__ void __launch_bounds__(1024, 1)
lstm_scan_kernel(const __bf16* __restrict__ xp_fw,
                 const __bf16* __restrict__ xp_rv,
                 const __bf16* __restrict__ WhT_fw,
                 const __bf16* __restrict__ WhT_rv,
                 const float* __restrict__ b_fw,
                 const float* __restrict__ b_rv,
                 float* __restrict__ out) {
    const int dir = blockIdx.x;
    const __bf16*  xp   = dir ? xp_rv  : xp_fw;
    const __bf16*  WhT  = dir ? WhT_rv : WhT_fw;
    const float*   bias = dir ? b_rv   : b_fw;

    extern __shared__ char smem[];
    float*  z_s    = (float*)smem;                 // [16][1024]  64 KB
    float*  c_s    = z_s + BATCH * GATE;           // [16][256]   16 KB
    float*  hf_s   = c_s + BATCH * HID;            // [16][256]   16 KB
    float*  bias_s = hf_s + BATCH * HID;           // [1024]       4 KB
    __bf16* h_s    = (__bf16*)(bias_s + GATE);     // [16][256] bf16 8 KB

    const int tid  = threadIdx.x;
    const int wave = tid >> 5;
    const int lane = tid & 31;
    const int hi   = (lane >> 4) & 1;
    const int lm   = lane & 15;

    // ---- init state ----
    if (tid < GATE) bias_s[tid] = bias[tid];
    for (int e = tid; e < BATCH * HID; e += 1024) {
        c_s[e] = 0.0f; hf_s[e] = 0.0f; h_s[e] = (__bf16)0.0f;
    }
    __syncthreads();

    // ---- resident B fragments for ntile0; row pointers for streamed ntile1 ----
    const int n0 = wave * 32 + lm;          // ntile0 column
    const int n1 = n0 + 16;                 // ntile1 column
    const uint4* w0row = (const uint4*)(WhT + (size_t)n0 * HID);
    const uint4* w1row = (const uint4*)(WhT + (size_t)n1 * HID);

    Frag Bf0[8];
#pragma unroll
    for (int kt = 0; kt < 8; ++kt) {
        Bf0[kt].q[0] = w0row[kt * 4 + hi];
        Bf0[kt].q[1] = w0row[kt * 4 + hi + 2];
    }

    const uint4* hrow = (const uint4*)(h_s + lm * HID);    // A row in LDS (32 uint4)

    // ---- time loop ----
    for (int t = 0; t < SEQ; ++t) {
        floatx8 acc0 = {0.f, 0.f, 0.f, 0.f, 0.f, 0.f, 0.f, 0.f};
        floatx8 acc1 = {0.f, 0.f, 0.f, 0.f, 0.f, 0.f, 0.f, 0.f};

#pragma unroll
        for (int kt = 0; kt < 8; ++kt) {
            Frag a;                                 // A = h (16x32 bf16 slice) from LDS
            a.q[0] = hrow[kt * 4 + hi];
            a.q[1] = hrow[kt * 4 + hi + 2];

            Frag b1;                                // streamed ntile1 B fragment (L2 hit)
            b1.q[0] = w1row[kt * 4 + hi];
            b1.q[1] = w1row[kt * 4 + hi + 2];

            acc0 = WMMA_BF16(a.v, Bf0[kt].v, acc0);
            acc1 = WMMA_BF16(a.v, b1.v, acc1);
        }

        // z = Wh-part + xp[t] + bias  -> LDS
        const __bf16* xpt = xp + (size_t)t * BATCH * GATE;
#pragma unroll
        for (int r = 0; r < 8; ++r) {
            const int m = r + hi * 8;
            z_s[m * GATE + n0] = acc0[r] + (float)xpt[m * GATE + n0] + bias_s[n0];
            z_s[m * GATE + n1] = acc1[r] + (float)xpt[m * GATE + n1] + bias_s[n1];
        }
        if (t + 1 < SEQ)
            __builtin_prefetch(xp + (size_t)(t + 1) * BATCH * GATE + (size_t)hi * 8 * GATE + n0, 0, 1);
        __syncthreads();

        // ---- gates & state update: 1024 threads x 4 features ----
        const int m2 = tid >> 6;                 // batch row 0..15
        const int j0 = (tid & 63) * 4;           // feature base 0..252
        const float4 zi4 = *(const float4*)(z_s + m2 * GATE + j0);
        const float4 zf4 = *(const float4*)(z_s + m2 * GATE + 256 + j0);
        const float4 zg4 = *(const float4*)(z_s + m2 * GATE + 512 + j0);
        const float4 zo4 = *(const float4*)(z_s + m2 * GATE + 768 + j0);
        float4 c4 = *(const float4*)(c_s + m2 * HID + j0);

        float4 h4;
        {
            float c;
            c = sigmoidf_(zf4.x) * c4.x + sigmoidf_(zi4.x) * tanhf_(zg4.x); c4.x = c; h4.x = sigmoidf_(zo4.x) * tanhf_(c);
            c = sigmoidf_(zf4.y) * c4.y + sigmoidf_(zi4.y) * tanhf_(zg4.y); c4.y = c; h4.y = sigmoidf_(zo4.y) * tanhf_(c);
            c = sigmoidf_(zf4.z) * c4.z + sigmoidf_(zi4.z) * tanhf_(zg4.z); c4.z = c; h4.z = sigmoidf_(zo4.z) * tanhf_(c);
            c = sigmoidf_(zf4.w) * c4.w + sigmoidf_(zi4.w) * tanhf_(zg4.w); c4.w = c; h4.w = sigmoidf_(zo4.w) * tanhf_(c);
        }
        *(float4*)(c_s + m2 * HID + j0)  = c4;
        *(float4*)(hf_s + m2 * HID + j0) = h4;
        h_s[m2 * HID + j0 + 0] = (__bf16)h4.x;
        h_s[m2 * HID + j0 + 1] = (__bf16)h4.y;
        h_s[m2 * HID + j0 + 2] = (__bf16)h4.z;
        h_s[m2 * HID + j0 + 3] = (__bf16)h4.w;

        // stream h to output [B, S, 2F]
        const size_t sOut = dir ? (size_t)(SEQ - 1 - t) : (size_t)t;
        *(float4*)(out + ((size_t)m2 * SEQ + sOut) * 512 + (size_t)dir * 256 + j0) = h4;

        __syncthreads();
    }

    // ---- final carries: (c_fw, h_fw, c_rv, h_rv) appended after the output ----
    const size_t base = (size_t)BATCH * SEQ * 512;
    {
        const int m = tid >> 6;
        const int j = (tid & 63) * 4;
        *(float4*)(out + base + (size_t)dir * 8192 + m * HID + j) =
            *(const float4*)(c_s + m * HID + j);
        *(float4*)(out + base + 4096 + (size_t)dir * 8192 + m * HID + j) =
            *(const float4*)(hf_s + m * HID + j);
    }
}

// ---------------------------------------------------------------------------
// Host launcher
// ---------------------------------------------------------------------------
extern "C" void kernel_launch(void* const* d_in, const int* in_sizes, int n_in,
                              void* d_out, int out_size, void* d_ws, size_t ws_size,
                              hipStream_t stream) {
    (void)in_sizes; (void)n_in; (void)out_size; (void)ws_size;

    const float* x     = (const float*)d_in[0];
    const float* Wi_fw = (const float*)d_in[1];
    const float* Wh_fw = (const float*)d_in[2];
    const float* b_fw  = (const float*)d_in[3];
    const float* Wi_rv = (const float*)d_in[4];
    const float* Wh_rv = (const float*)d_in[5];
    const float* b_rv  = (const float*)d_in[6];
    float* out = (float*)d_out;

    // workspace layout (bf16 staging)
    const size_t XP_ELEMS = (size_t)SEQ * BATCH * GATE;    // 67,108,864
    const size_t WT_ELEMS = (size_t)GATE * HID;            // 262,144
    const size_t X_ELEMS  = (size_t)BATCH * SEQ * HID;     // 16,777,216
    __bf16* xp_fw  = (__bf16*)d_ws;
    __bf16* xp_rv  = xp_fw + XP_ELEMS;
    __bf16* WiT_fw = xp_rv + XP_ELEMS;
    __bf16* WhT_fw = WiT_fw + WT_ELEMS;
    __bf16* WiT_rv = WhT_fw + WT_ELEMS;
    __bf16* WhT_rv = WiT_rv + WT_ELEMS;
    __bf16* xbf    = WhT_rv + WT_ELEMS;

    // 0) stage x as bf16
    pack_x_kernel<<<(unsigned)(X_ELEMS / 4 / 256), 256, 0, stream>>>(x, xbf);

    // 1) pack/transpose weights to bf16 [N][K]
    pack_weights_kernel<<<4096, 256, 0, stream>>>(Wi_fw, Wh_fw, Wi_rv, Wh_rv,
                                                  WiT_fw, WhT_fw, WiT_rv, WhT_rv);

    // 2) pre-projection GEMM for both directions (bf16 WMMA)
    xproj_gemm_kernel<<<dim3(8, SEQ, 2), 256, 0, stream>>>(xbf, WiT_fw, WiT_rv, xp_fw, xp_rv);

    // 3) persistent recurrent scan: one workgroup per direction
    const size_t shmem = (size_t)(BATCH * GATE + BATCH * HID * 2 + GATE) * sizeof(float)
                       + (size_t)BATCH * HID * sizeof(__bf16);   // 108 KB
    lstm_scan_kernel<<<2, 1024, shmem, stream>>>(xp_fw, xp_rv, WhT_fw, WhT_rv,
                                                 b_fw, b_rv, out);
}
